// Block_12240656794281
// MI455X (gfx1250) — matrix operand order repaired and checked by reading, not compiled, not run on hardware
//
#include <hip/hip_runtime.h>
#include <hip/hip_bf16.h>
#include <math.h>

// ---------------------------------------------------------------------------
// Transformer block for MI455X (gfx1250): wave32 + V_WMMA_F32_16X16X32_F16.
// GEMMs: all fragments are contiguous global b128 loads (weights pre-transposed
// to [N,K] f16), zero LDS, zero barriers -> pure load/wmma stream out of L2.
// Flash attention: V is kept transposed ([B,H,D,T] f16, produced by the V-proj
// epilogue); the Tensor Data Mover stages a [64 x 32] V^T tile into LDS
// (s_wait_tensorcnt), and every fragment read is a contiguous b128/ds_b128.
// ---------------------------------------------------------------------------

typedef _Float16 f16;
typedef __attribute__((ext_vector_type(16))) _Float16 v16h;
typedef __attribute__((ext_vector_type(8)))  _Float16 v8h;
typedef __attribute__((ext_vector_type(8)))  float    v8f;
typedef __attribute__((ext_vector_type(4)))  unsigned int u32x4;
typedef __attribute__((ext_vector_type(8)))  int      i32x8;
typedef __attribute__((ext_vector_type(4)))  int      i32x4;

#define N_EMBD 1024
#define N_HEAD 16
#define HDIM   64
#define SEQ    2048
#define BATCH  2
#define BT     (BATCH * SEQ)        // 4096 rows

#if defined(__has_builtin)
#if __has_builtin(__builtin_amdgcn_tensor_load_to_lds)
#define HAS_TDM 1
#endif
#endif
#ifndef HAS_TDM
#define HAS_TDM 0
#endif

// ---------------- WMMA wrapper ----------------------------------------------
__device__ __forceinline__ v8f wmma16(v16h a, v16h b, v8f c) {
  // (neg_a, A, neg_b, B, c_mod, C, reuse_a, reuse_b)
  return __builtin_amdgcn_wmma_f32_16x16x32_f16(false, a, false, b,
                                                (short)0, c, false, false);
}

__device__ __forceinline__ v16h cat16(v8h lo, v8h hi) {
  return __builtin_shufflevector(lo, hi, 0, 1, 2, 3, 4, 5, 6, 7,
                                 8, 9, 10, 11, 12, 13, 14, 15);
}

// ---------------- Fragment loaders (ISA 7.12.2 layouts, wave32) -------------
// A-matrix 16x32 f16 from row-major memory (row stride in elements):
// lane<16 -> row=lane; kh=lane>>4; K = {kh*8+0..7, kh*8+16..23}
__device__ __forceinline__ v16h frag_a_g(const f16* base, long stride) {
  int lane = threadIdx.x & 31;
  const f16* p = base + (long)(lane & 15) * stride + (lane >> 4) * 8;
  return cat16(*(const v8h*)p, *(const v8h*)(p + 16));
}

// B-matrix 32x16 fragment from column-major storage (i.e. row n of W^T[N,K]):
// lane holds col n = lane&15 (caller offsets base); K = {kh*8+0..7, kh*8+16..23}
// (kh=lane>>4) -> two b128 loads, same K order as frag_a_g.
__device__ __forceinline__ v16h frag_b_ct(const f16* colbase, long stride) {
  int lane = threadIdx.x & 31;
  const f16* p = colbase + (long)(lane & 15) * stride + (lane >> 4) * 8;
  return cat16(*(const v8h*)p, *(const v8h*)(p + 16));
}

// ---------------- TDM helper -------------------------------------------------
#if HAS_TDM
// 2-D tensor tile load: global (row-major, dim0 contiguous) -> LDS, f16 data.
__device__ __forceinline__ void tdm_load_2d(unsigned lds_byte_addr,
                                            const void* gtile,
                                            unsigned tile_d0, unsigned tile_d1,
                                            unsigned tensor_d0, unsigned tensor_d1,
                                            unsigned stride_d0 /*elements*/) {
  unsigned long long ga = (unsigned long long)gtile;
  u32x4 g0;
  g0[0] = 1u;                                         // count=1 (user descriptor)
  g0[1] = lds_byte_addr;                              // LDS byte address
  g0[2] = (unsigned)(ga & 0xffffffffu);               // global_addr[31:0]
  g0[3] = (unsigned)((ga >> 32) & 0x1ffffffu)         // global_addr[56:32]
        | (2u << 30);                                 // type = 2 ("image")
  i32x8 g1;
  g1[0] = (int)(1u << 16);                            // data_size = 1 -> 2 bytes
  g1[1] = (int)((tensor_d0 & 0xffffu) << 16);         // tensor_dim0[15:0]
  g1[2] = (int)((tensor_d0 >> 16) | ((tensor_d1 & 0xffffu) << 16));
  g1[3] = (int)((tensor_d1 >> 16) | (tile_d0 << 16)); // tile_dim0
  g1[4] = (int)(tile_d1 & 0xffffu);                   // tile_dim1 (tile_dim2=0)
  g1[5] = (int)stride_d0;                             // tensor_dim0_stride[31:0]
  g1[6] = 0;
  g1[7] = 0;
  i32x4 z4 = {0, 0, 0, 0};
#if __clang_major__ >= 23
  i32x8 z8 = {0, 0, 0, 0, 0, 0, 0, 0};
  __builtin_amdgcn_tensor_load_to_lds(g0, g1, z4, z4, z8, 0);
#else
  __builtin_amdgcn_tensor_load_to_lds(g0, g1, z4, z4, 0);
#endif
}
#endif

// ---------------- Elementwise kernels ---------------------------------------
__global__ void add_f32_kernel(const float* __restrict__ a,
                               const float* __restrict__ b,
                               float* __restrict__ out, long n) {
  long i = (long)blockIdx.x * blockDim.x + threadIdx.x;
  if (i < n) out[i] = a[i] + b[i];
}

// Transpose fp32 [K,N] -> f16 [N,K] (32x32 LDS tiles, 256 threads).
__global__ __launch_bounds__(256)
void transpose_f32_to_f16_kernel(const float* __restrict__ in,
                                 f16* __restrict__ out, int K, int N) {
  __shared__ float tile[32][33];
  int n0 = blockIdx.x * 32, k0 = blockIdx.y * 32;
  int tx = threadIdx.x & 31, ty = threadIdx.x >> 5;   // 32 x 8
#pragma unroll
  for (int i = 0; i < 4; ++i)
    tile[ty + 8 * i][tx] = in[(long)(k0 + ty + 8 * i) * N + n0 + tx];
  __syncthreads();
#pragma unroll
  for (int i = 0; i < 4; ++i)
    out[(long)(n0 + ty + 8 * i) * K + k0 + tx] = (f16)tile[tx][ty + 8 * i];
}

// LayerNorm over C=1024, one row per block (256 threads), f16 output.
__global__ __launch_bounds__(256)
void ln_f16_kernel(const float* __restrict__ x, const float* __restrict__ w,
                   const float* __restrict__ b, f16* __restrict__ out) {
  const int C = N_EMBD;
  long row = blockIdx.x;
  const float* xr = x + row * C;
  int tid = threadIdx.x;
  float vals[4], s = 0.f, ss = 0.f;
#pragma unroll
  for (int i = 0; i < 4; ++i) {
    float v = xr[tid + i * 256];
    vals[i] = v; s += v; ss += v * v;
  }
#pragma unroll
  for (int off = 16; off >= 1; off >>= 1) {
    s  += __shfl_xor(s, off, 32);
    ss += __shfl_xor(ss, off, 32);
  }
  __shared__ float sm[8], sm2[8];
  if ((tid & 31) == 0) { sm[tid >> 5] = s; sm2[tid >> 5] = ss; }
  __syncthreads();
  s = 0.f; ss = 0.f;
#pragma unroll
  for (int i = 0; i < 8; ++i) { s += sm[i]; ss += sm2[i]; }
  float mu  = s / C;
  float var = ss / C - mu * mu;
  float r   = rsqrtf(var + 1e-5f);
#pragma unroll
  for (int i = 0; i < 4; ++i) {
    int c = tid + i * 256;
    out[row * C + c] = (f16)((vals[i] - mu) * r * w[c] + b[c]);
  }
}

// ---------------- WMMA GEMM (no LDS, no barriers) ----------------------------
// C[M,N] = A[M,K] @ W[K,N] + bias with W pre-transposed as WT[N,K] f16.
// Block: 128 threads (4 waves); tile 64(M)x64(N); wave w -> rows w*16..+16.
// Fragments are contiguous b128 global loads, reuse served by the 192MB L2.
// EPI 0: qkv scatter -> fp32 [B,H,T,D] (if dstF) + f16 copy (dstH, same layout)
//        + optional transposed f16 copy [B,H,D,T] (dstHT) for V.
// EPI 1: exact GELU -> f16 row-major [M,N] (dstH)
// EPI 2: residual add -> fp32 row-major [M,N] (dstF), res same layout
template <int EPI>
__global__ __launch_bounds__(128)
void gemm_f16_kernel(const f16* __restrict__ A, const f16* __restrict__ WT,
                     const float* __restrict__ bias,
                     const float* __restrict__ res,
                     float* __restrict__ dstF, f16* __restrict__ dstH,
                     f16* __restrict__ dstHT,
                     int M, int N, int K) {
  int tid = threadIdx.x;
  int wave = tid >> 5, lane = tid & 31;
  int m0 = blockIdx.y * 64 + wave * 16;
  int n0 = blockIdx.x * 64;
  const f16* arow = A + (long)m0 * K;

  v8f acc[4];
#pragma unroll
  for (int j = 0; j < 4; ++j)
#pragma unroll
    for (int i = 0; i < 8; ++i) acc[j][i] = 0.f;

  for (int k0 = 0; k0 < K; k0 += 32) {
    v16h a = frag_a_g(arow + k0, K);
#pragma unroll
    for (int j = 0; j < 4; ++j) {
      v16h b = frag_b_ct(WT + (long)(n0 + j * 16) * K + k0, K);
      acc[j] = wmma16(a, b, acc[j]);
    }
    if (k0 + 32 < K) {  // hint next K-chunk into cache (global_prefetch_b8)
      __builtin_prefetch(arow + (long)(lane & 15) * K + k0 + 32, 0, 1);
      __builtin_prefetch(WT + (long)(n0 + (lane & 15)) * K + k0 + 32, 0, 1);
    }
  }

  // Epilogue (C/D layout: lane<16 -> col=lane,row=vi ; lane>=16 -> row=vi+8)
  int half = lane >> 4, nl = lane & 15;
#pragma unroll
  for (int j = 0; j < 4; ++j) {
    int n = n0 + j * 16 + nl;
    float bb = bias[n];
#pragma unroll
    for (int vi = 0; vi < 8; ++vi) {
      int m = m0 + vi + 8 * half;
      float val = acc[j][vi] + bb;
      if (EPI == 0) {
        int bb_ = m >> 11, t = m & (SEQ - 1);          // T = 2048
        int h = n >> 6, d = n & (HDIM - 1);
        long bh = (long)(bb_ * N_HEAD + h);
        long idx = (bh * SEQ + t) * HDIM + d;
        if (dstF) dstF[idx] = val;
        dstH[idx] = (f16)val;
        if (dstHT) dstHT[(bh * HDIM + d) * SEQ + t] = (f16)val;
      } else if (EPI == 1) {
        float g = 0.5f * val * (1.f + erff(val * 0.70710678118654752f));
        dstH[(long)m * N + n] = (f16)g;
      } else {
        dstF[(long)m * N + n] = res[(long)m * N + n] + val;
      }
    }
  }
}

// ---------------- Flash-style causal attention ------------------------------
// Grid: (T/64, B*H). Block: 128 threads (4 waves), wave owns 16 query rows.
// S = Q K^T / 8 with online softmax, P@V via WMMA.
// V^T [B,H,D,T] tile (64 x 32) staged into LDS by TDM; all fragment reads are
// contiguous (global b128 / ds_load_b128).
__global__ __launch_bounds__(128)
void flash_attn_kernel(const f16* __restrict__ q, const f16* __restrict__ k,
                       const f16* __restrict__ vT, float* __restrict__ y) {
  __shared__ f16 Pld[4 * 16 * 32];      // per-wave P tile [16 x 32]
  __shared__ f16 VldT[64 * 32];         // V^T chunk: [d 0..63][t-chunk 0..31]
  int tid = threadIdx.x, wave = tid >> 5, lane = tid & 31;
  int half = lane >> 4, nl = lane & 15;

  int bh = blockIdx.y;                  // b*H + h
  const f16* qb  = q  + (long)bh * SEQ * HDIM;
  const f16* kb  = k  + (long)bh * SEQ * HDIM;
  const f16* vTb = vT + (long)bh * HDIM * SEQ;
  int b = bh >> 4, h = bh & (N_HEAD - 1);

  int r0 = blockIdx.x * 64 + wave * 16; // wave's first query row

  v16h qf0 = frag_a_g(qb + (long)r0 * HDIM,      HDIM);
  v16h qf1 = frag_a_g(qb + (long)r0 * HDIM + 32, HDIM);

  float m8[8], l8[8];
  v8f yacc[4];
#pragma unroll
  for (int i = 0; i < 8; ++i) { m8[i] = -INFINITY; l8[i] = 0.f; }
#pragma unroll
  for (int j = 0; j < 4; ++j)
#pragma unroll
    for (int i = 0; i < 8; ++i) yacc[j][i] = 0.f;

  int s_end = blockIdx.x * 64 + 64;     // causal upper bound for this block
  for (int s0 = 0; s0 < s_end; s0 += 32) {
    // Stage V^T[0:64, s0:s0+32] into LDS
#if HAS_TDM
    if (wave == 0)
      tdm_load_2d((unsigned)(unsigned long long)(void*)VldT,
                  vTb + s0,
                  /*tile d0,d1*/ 32, HDIM, /*tensor d0,d1*/ SEQ, HDIM,
                  /*stride*/ SEQ);
#else
#pragma unroll
    for (int it = 0; it < 2; ++it) {
      int idx = tid * 2 + it;           // 0..255
      int r = idx >> 2, c8 = (idx & 3) * 8;  // 64 rows x 32 cols
      *(v8h*)&VldT[r * 32 + c8] =
          *(const v8h*)&vTb[(long)r * SEQ + s0 + c8];
    }
#endif

    // Two 16x16 S tiles (keys s0..s0+16, s0+16..s0+32), K-dim = D = 64
    v8f sA, sB;
#pragma unroll
    for (int i = 0; i < 8; ++i) { sA[i] = 0.f; sB[i] = 0.f; }
    sA = wmma16(qf0, frag_b_ct(kb + (long)s0 * HDIM,          HDIM), sA);
    sA = wmma16(qf1, frag_b_ct(kb + (long)s0 * HDIM + 32,     HDIM), sA);
    sB = wmma16(qf0, frag_b_ct(kb + (long)(s0 + 16) * HDIM,     HDIM), sB);
    sB = wmma16(qf1, frag_b_ct(kb + (long)(s0 + 16) * HDIM + 32, HDIM), sB);

    const float scale = 0.125f;         // 1/sqrt(64)
    float alpha[8];
    f16* Pw = &Pld[wave * 16 * 32];
#pragma unroll
    for (int vi = 0; vi < 8; ++vi) {
      int tq = r0 + vi + 8 * half;
      float a = (s0 + nl      <= tq) ? sA[vi] * scale : -INFINITY;
      float c = (s0 + 16 + nl <= tq) ? sB[vi] * scale : -INFINITY;
      float mx = fmaxf(a, c);
#pragma unroll
      for (int off = 1; off < 16; off <<= 1) mx = fmaxf(mx, __shfl_xor(mx, off, 16));
      float mn = fmaxf(m8[vi], mx);
      alpha[vi] = __expf(m8[vi] - mn);
      float pa = __expf(a - mn);
      float pb = __expf(c - mn);
      float rs = pa + pb;
#pragma unroll
      for (int off = 1; off < 16; off <<= 1) rs += __shfl_xor(rs, off, 16);
      l8[vi] = l8[vi] * alpha[vi] + rs;
      m8[vi] = mn;
      int row = vi + 8 * half;
      Pw[row * 32 + nl]      = (f16)pa;
      Pw[row * 32 + 16 + nl] = (f16)pb;
    }
#if HAS_TDM
    if (wave == 0) __builtin_amdgcn_s_wait_tensorcnt(0);
#endif
    __syncthreads();

    // y = y*alpha + P @ V  (K-dim = 32 keys); V^T tile -> contiguous ds_b128
    v16h pf = frag_a_g(Pw, 32);
#pragma unroll
    for (int j = 0; j < 4; ++j) {
      v8f cprev;
#pragma unroll
      for (int vi = 0; vi < 8; ++vi) cprev[vi] = yacc[j][vi] * alpha[vi];
      v16h vf = frag_b_ct(&VldT[(j * 16) * 32], 32);
      yacc[j] = wmma16(pf, vf, cprev);
    }
    __syncthreads();
  }

  // y[b, t, h*64 + d] = yacc / l
#pragma unroll
  for (int j = 0; j < 4; ++j)
#pragma unroll
    for (int vi = 0; vi < 8; ++vi) {
      int t = r0 + vi + 8 * half;
      int d = j * 16 + nl;
      y[((long)(b * SEQ + t)) * N_EMBD + h * HDIM + d] = yacc[j][vi] / l8[vi];
    }
}

// ---------------------------------------------------------------------------
extern "C" void kernel_launch(void* const* d_in, const int* in_sizes, int n_in,
                              void* d_out, int out_size, void* d_ws, size_t ws_size,
                              hipStream_t stream) {
  (void)in_sizes; (void)n_in; (void)out_size; (void)ws_size;
  const float* x     = (const float*)d_in[0];
  const float* wq    = (const float*)d_in[1];
  const float* bq    = (const float*)d_in[2];
  const float* wk    = (const float*)d_in[3];
  const float* bk    = (const float*)d_in[4];
  const float* wv    = (const float*)d_in[5];
  const float* bv    = (const float*)d_in[6];
  const float* ln1w  = (const float*)d_in[7];
  const float* ln1b  = (const float*)d_in[8];
  const float* ln2w  = (const float*)d_in[9];
  const float* ln2b  = (const float*)d_in[10];
  const float* wfc   = (const float*)d_in[11];
  const float* bfc   = (const float*)d_in[12];
  const float* wproj = (const float*)d_in[13];
  const float* bproj = (const float*)d_in[14];

  const long CC  = (long)N_EMBD * N_EMBD;       // 1M
  const long C4  = (long)N_EMBD * 4 * N_EMBD;   // 4M
  const long BTC = (long)BT * N_EMBD;           // 4M

  float* outX = (float*)d_out;
  float* outK = outX + BTC;
  float* outV = outK + BTC;

  // Workspace carve-out
  char* wp = (char*)d_ws;
  auto alloc = [&](size_t bytes) {
    char* p = wp; wp += (bytes + 255) & ~(size_t)255; return p;
  };
  f16*   wqT   = (f16*)alloc(CC * 2);   // [C,C]   (N,K)
  f16*   wkT   = (f16*)alloc(CC * 2);
  f16*   wvT   = (f16*)alloc(CC * 2);
  f16*   wfcT  = (f16*)alloc(C4 * 2);   // [4C, C]
  f16*   wprT  = (f16*)alloc(C4 * 2);   // [C, 4C]
  f16*   h1    = (f16*)alloc(BTC * 2);
  f16*   q16   = (f16*)alloc(BTC * 2);
  f16*   k16   = (f16*)alloc(BTC * 2);
  f16*   vT16  = (f16*)alloc(BTC * 2);  // V transposed: [B,H,D,T]
  float* ybuf  = (float*)alloc(BTC * 4);
  float* x1    = (float*)alloc(BTC * 4);
  f16*   h2    = (f16*)alloc(BTC * 2);
  f16*   fc16  = (f16*)alloc((long)BT * 4 * N_EMBD * 2);

  // 1) weight transposes fp32 [K,N] -> f16 [N,K]
  dim3 tb(256);
  transpose_f32_to_f16_kernel<<<dim3(N_EMBD / 32, N_EMBD / 32), tb, 0, stream>>>(
      wq, wqT, N_EMBD, N_EMBD);
  transpose_f32_to_f16_kernel<<<dim3(N_EMBD / 32, N_EMBD / 32), tb, 0, stream>>>(
      wk, wkT, N_EMBD, N_EMBD);
  transpose_f32_to_f16_kernel<<<dim3(N_EMBD / 32, N_EMBD / 32), tb, 0, stream>>>(
      wv, wvT, N_EMBD, N_EMBD);
  transpose_f32_to_f16_kernel<<<dim3(4 * N_EMBD / 32, N_EMBD / 32), tb, 0, stream>>>(
      wfc, wfcT, N_EMBD, 4 * N_EMBD);
  transpose_f32_to_f16_kernel<<<dim3(N_EMBD / 32, 4 * N_EMBD / 32), tb, 0, stream>>>(
      wproj, wprT, 4 * N_EMBD, N_EMBD);

  // 2) LN1 -> h1 (f16)
  ln_f16_kernel<<<BT, 256, 0, stream>>>(x, ln1w, ln1b, h1);

  // 3) QKV projections (WMMA GEMM, EPI 0: scatter to [B,H,T,D])
  dim3 gq(N_EMBD / 64, BT / 64);        // (16, 64)
  gemm_f16_kernel<0><<<gq, 128, 0, stream>>>(h1, wqT, bq, nullptr,
                                             nullptr, q16, nullptr,
                                             BT, N_EMBD, N_EMBD);
  gemm_f16_kernel<0><<<gq, 128, 0, stream>>>(h1, wkT, bk, nullptr,
                                             outK, k16, nullptr,
                                             BT, N_EMBD, N_EMBD);
  // V: fp32 -> d_out, f16 transposed copy [B,H,D,T] for the P@V WMMA
  // (dstH is required by the epilogue; reuse a scratch f16 buffer for it)
  gemm_f16_kernel<0><<<gq, 128, 0, stream>>>(h1, wvT, bv, nullptr,
                                             outV, h2 /*scratch*/, vT16,
                                             BT, N_EMBD, N_EMBD);

  // 4) causal flash attention -> ybuf (fp32 [B,T,C])
  dim3 ga(SEQ / 64, BATCH * N_HEAD);    // (32, 32)
  flash_attn_kernel<<<ga, 128, 0, stream>>>(q16, k16, vT16, ybuf);

  // 5) residual: x1 = x + y
  add_f32_kernel<<<(BTC + 255) / 256, 256, 0, stream>>>(x, ybuf, x1, BTC);

  // 6) LN2 -> h2 (f16)
  ln_f16_kernel<<<BT, 256, 0, stream>>>(x1, ln2w, ln2b, h2);

  // 7) MLP fc: gelu(h2 @ wfc + bfc) -> fc16 (f16 [BT, 4C])
  dim3 gf(4 * N_EMBD / 64, BT / 64);    // (64, 64)
  gemm_f16_kernel<1><<<gf, 128, 0, stream>>>(h2, wfcT, bfc, nullptr,
                                             nullptr, fc16, nullptr,
                                             BT, 4 * N_EMBD, N_EMBD);

  // 8) MLP proj + residual: outX = x1 + fc16 @ wproj + bproj
  gemm_f16_kernel<2><<<gq, 128, 0, stream>>>(fc16, wprT, bproj, x1,
                                             outX, nullptr, nullptr,
                                             BT, N_EMBD, 4 * N_EMBD);
}
